// CausalSelfAttention_35467839930630
// MI455X (gfx1250) — compile-verified, matrix operand order checked by reading, not run on hardware
//
#include <hip/hip_runtime.h>
#include <hip/hip_bf16.h>

// Problem constants
#define Bsz  4
#define Tseq 2048
#define Cdim 1024
#define Hn   16
#define HDim 64
#define N3C  3072

typedef __attribute__((ext_vector_type(16))) __bf16 v16bf;
typedef __attribute__((ext_vector_type(8)))  float  v8f;
typedef unsigned int v4u __attribute__((ext_vector_type(4)));
typedef unsigned int v8u __attribute__((ext_vector_type(8)));

union FragU { uint4 u[2]; v16bf v; };

__device__ __forceinline__ unsigned short f2bf(float f) {
    union { float f; unsigned u; } a; a.f = f;
    unsigned r = a.u + 0x7FFFu + ((a.u >> 16) & 1u);
    return (unsigned short)(r >> 16);
}

// ---------------- prep kernels ----------------
__global__ void k_cvt(const float* __restrict__ in, unsigned short* __restrict__ out, int n) {
    int i = blockIdx.x * blockDim.x + threadIdx.x;
    if (i < n) out[i] = f2bf(in[i]);
}

// in: [K,N] fp32 -> out: [N,K] bf16
__global__ void k_transpose_cvt(const float* __restrict__ in, unsigned short* __restrict__ out,
                                int K, int N) {
    int i = blockIdx.x * blockDim.x + threadIdx.x;
    if (i < K * N) {
        int n = i / K, k = i - n * K;
        out[i] = f2bf(in[k * N + n]);
    }
}

// Issue one TDM descriptor: load a 128-row x 64-col bf16 tile (row stride K elems)
// into LDS at ldsAddr, with 16B padding inserted after every 128B row
// (pad_interval=4 -> 32 DWORDs, pad_amount=3 -> 4 DWORDs) => LDS row stride 144B.
__device__ __forceinline__ void tdm_load_tile(unsigned ldsAddr,
                                              const unsigned short* gptr, int K) {
    unsigned long long ga = (unsigned long long)(size_t)gptr;
    v4u g0;
    g0[0] = 1u;                                    // count=1, user mode
    g0[1] = ldsAddr;                               // lds_addr [63:32]
    g0[2] = (unsigned)ga;                          // global_addr [95:64]
    g0[3] = (unsigned)(ga >> 32) | 0x80000000u;    // global_addr hi | type=2
    v8u g1;
    g1[0] = (1u << 16)      // data_size = 2 bytes
          | (1u << 20)      // pad_enable
          | (4u << 22)      // pad_interval: 32 DWORDs (one 128B row)
          | (3u << 25);     // pad_amount: 4 DWORDs (16B)
    g1[1] = ((unsigned)K & 0xFFFFu) << 16;         // tensor_dim0[15:0]
    g1[2] = ((unsigned)K >> 16) | (128u << 16);    // tensor_dim0[31:16] | tensor_dim1 lo
    g1[3] = (64u << 16);                           // tensor_dim1 hi | tile_dim0=64
    g1[4] = 128u;                                  // tile_dim1=128, tile_dim2=0
    g1[5] = (unsigned)K;                           // tensor_dim0_stride lo32 (elems)
    g1[6] = 0u;
    g1[7] = 0u;
    v4u gz; gz[0] = 0u; gz[1] = 0u; gz[2] = 0u; gz[3] = 0u;
    asm volatile("tensor_load_to_lds %0, %1, %2, %3"
                 :: "s"(g0), "s"(g1), "s"(gz), "s"(gz) : "memory");
}

// ---------------- tiled WMMA GEMM (TDM double-buffered) ----------------
// C[M,N] = A[M,K] * BT[N,K]^T + bias ; MODE 0: scatter to q/k/vT bf16, MODE 1: fp32 out
#define LROW 72   // padded LDS row stride in elements (144B)
template<int MODE>
__global__ __launch_bounds__(256)
void k_gemm(const unsigned short* __restrict__ A,
            const unsigned short* __restrict__ BT,
            const float* __restrict__ bias,
            int M, int N, int K,
            unsigned short* __restrict__ q, unsigned short* __restrict__ kbuf,
            unsigned short* __restrict__ vT, float* __restrict__ out)
{
    __shared__ alignas(16) unsigned short As[2][128 * LROW];
    __shared__ alignas(16) unsigned short Bs[2][128 * LROW];
    const int tid  = threadIdx.x;
    const int lane = tid & 31;
    const int w    = tid >> 5;
    const int wm   = w & 3;              // 4 waves along M
    const int wn   = w >> 2;             // 2 waves along N
    const int l16  = lane & 15;
    const int half = (lane >> 4) << 3;   // element offset 0 or 8
    const int rofs = (lane >> 4) << 3;   // row offset 0 or 8 for D layout
    const int bm   = blockIdx.x * 128;
    const int bn   = blockIdx.y * 128;

    v8f acc[2][4] = {};

    const int nk = K >> 6;   // 64-wide K chunks
    if (w == 0) {
        tdm_load_tile((unsigned)(size_t)&As[0][0], A  + (size_t)bm * K, K);
        tdm_load_tile((unsigned)(size_t)&Bs[0][0], BT + (size_t)bn * K, K);
    }

    for (int i = 0; i < nk; ++i) {
        const int cur = i & 1;
        if (w == 0) {
            if (i + 1 < nk) {
                // prefetch next chunk into the other buffer, then wait only for
                // the current buffer's two loads (TENSORcnt completes in order)
                tdm_load_tile((unsigned)(size_t)&As[cur ^ 1][0],
                              A + (size_t)bm * K + (i + 1) * 64, K);
                tdm_load_tile((unsigned)(size_t)&Bs[cur ^ 1][0],
                              BT + (size_t)bn * K + (i + 1) * 64, K);
                __builtin_amdgcn_s_wait_tensorcnt(2);
            } else {
                __builtin_amdgcn_s_wait_tensorcnt(0);
            }
        }
        __syncthreads();

        v16bf af[2][2], bfv[4][2];
        #pragma unroll
        for (int mt = 0; mt < 2; ++mt) {
            int row = wm * 32 + mt * 16 + l16;
            #pragma unroll
            for (int ks = 0; ks < 2; ++ks) {
                FragU f;
                f.u[0] = *(const uint4*)(&As[cur][row * LROW + ks * 32 + half]);
                f.u[1] = *(const uint4*)(&As[cur][row * LROW + ks * 32 + half + 16]);
                af[mt][ks] = f.v;
            }
        }
        #pragma unroll
        for (int nt = 0; nt < 4; ++nt) {
            int row = wn * 64 + nt * 16 + l16;
            #pragma unroll
            for (int ks = 0; ks < 2; ++ks) {
                FragU f;
                f.u[0] = *(const uint4*)(&Bs[cur][row * LROW + ks * 32 + half]);
                f.u[1] = *(const uint4*)(&Bs[cur][row * LROW + ks * 32 + half + 16]);
                bfv[nt][ks] = f.v;
            }
        }
        #pragma unroll
        for (int ks = 0; ks < 2; ++ks)
            #pragma unroll
            for (int mt = 0; mt < 2; ++mt)
                #pragma unroll
                for (int nt = 0; nt < 4; ++nt)
                    acc[mt][nt] = __builtin_amdgcn_wmma_f32_16x16x32_bf16(
                        false, af[mt][ks], false, bfv[nt][ks], (short)0,
                        acc[mt][nt], false, false);
        __syncthreads();
    }

    #pragma unroll
    for (int mt = 0; mt < 2; ++mt)
        #pragma unroll
        for (int nt = 0; nt < 4; ++nt) {
            int ncol  = bn + wn * 64 + nt * 16 + l16;
            float bvl = bias[ncol];
            #pragma unroll
            for (int r = 0; r < 8; ++r) {
                int   mrow = bm + wm * 32 + mt * 16 + r + rofs;
                float vv   = acc[mt][nt][r] + bvl;
                if (MODE == 0) {
                    int which = ncol >> 10;          // 0:q 1:k 2:v
                    int rem   = ncol & 1023;
                    int h     = rem >> 6;
                    int d     = rem & 63;
                    int b     = mrow >> 11;          // / Tseq
                    int t     = mrow & (Tseq - 1);
                    int bh    = b * Hn + h;
                    unsigned short bv = f2bf(vv);
                    if (which == 0)      q   [((size_t)bh * Tseq + t) * HDim + d] = bv;
                    else if (which == 1) kbuf[((size_t)bh * Tseq + t) * HDim + d] = bv;
                    else                 vT  [((size_t)bh * HDim + d) * Tseq + t] = bv;
                } else {
                    out[(size_t)mrow * N + ncol] = vv;
                }
            }
        }
}

// ---------------- flash attention (causal, online softmax) ----------------
__global__ __launch_bounds__(256)
void k_flash(const unsigned short* __restrict__ Q,
             const unsigned short* __restrict__ Kb,
             const unsigned short* __restrict__ VT,
             unsigned short* __restrict__ Y)
{
    __shared__ alignas(16) unsigned short pLds[8][16 * 80];
    const int tid   = threadIdx.x;
    const int lane  = tid & 31;
    const int w     = tid >> 5;
    const int l16   = lane & 15;
    const int halfB = (lane >> 4) * 16;   // byte offset 0/16 inside fragment row
    const int rofs  = (lane >> 4) * 8;    // D-layout row offset

    const int bh = blockIdx.y;
    const int qb = blockIdx.x * 128 + w * 16;

    const char* qbase = (const char*)(Q  + (size_t)bh * Tseq * HDim);
    const char* kbase = (const char*)(Kb + (size_t)bh * Tseq * HDim);
    const char* vbase = (const char*)(VT + (size_t)bh * HDim * Tseq);

    // Q fragments: rows qb..qb+15, dim chunks [0,32) and [32,64)
    v16bf qf[2];
    #pragma unroll
    for (int f = 0; f < 2; ++f) {
        FragU fu;
        const char* rp = qbase + (size_t)(qb + l16) * 128 + f * 64 + halfB;
        fu.u[0] = *(const uint4*)(rp);
        fu.u[1] = *(const uint4*)(rp + 32);
        qf[f] = fu.v;
    }

    v8f o[4] = {};
    float mst[8], lst[8];
    #pragma unroll
    for (int r = 0; r < 8; ++r) { mst[r] = -1e38f; lst[r] = 0.f; }

    const float sc  = 0.125f * 1.44269504088896f;   // 1/sqrt(64) * log2(e)
    const int  nkb  = blockIdx.x * 2 + 2;           // 64-key blocks (causal bound of WG)
    unsigned short* pw = pLds[w];

    for (int kb64 = 0; kb64 < nkb; ++kb64) {
        const int kb = kb64 * 64;

        // Load ALL K fragments first (lets loads overlap the matrix pipe)
        v16bf kf[4][2];
        #pragma unroll
        for (int nt = 0; nt < 4; ++nt)
            #pragma unroll
            for (int f = 0; f < 2; ++f) {
                FragU fu;
                const char* rp = kbase + (size_t)(kb + nt * 16 + l16) * 128 + f * 64 + halfB;
                fu.u[0] = *(const uint4*)(rp);
                fu.u[1] = *(const uint4*)(rp + 32);
                kf[nt][f] = fu.v;
            }
        // S = Q * K^T : 4 tiles of 16 keys, 2 k-steps of 32 dims
        v8f s[4] = {};
        #pragma unroll
        for (int f = 0; f < 2; ++f)
            #pragma unroll
            for (int nt = 0; nt < 4; ++nt)
                s[nt] = __builtin_amdgcn_wmma_f32_16x16x32_bf16(
                    false, qf[f], false, kf[nt][f], (short)0, s[nt], false, false);

        // online softmax per row (row r lives in VGPR r across one 16-lane half)
        #pragma unroll
        for (int r = 0; r < 8; ++r) {
            const int qrow = qb + r + rofs;
            float sv[4];
            float rmax = -1e30f;
            #pragma unroll
            for (int nt = 0; nt < 4; ++nt) {
                int   kidx = kb + nt * 16 + l16;
                float x    = s[nt][r] * sc;
                if (kidx > qrow) x = -1e30f;     // causal mask
                sv[nt] = x;
                rmax   = fmaxf(rmax, x);
            }
            #pragma unroll
            for (int msk = 1; msk < 16; msk <<= 1)
                rmax = fmaxf(rmax, __shfl_xor(rmax, msk, 32));
            float mnew  = fmaxf(mst[r], rmax);
            float alpha = exp2f(mst[r] - mnew);
            mst[r] = mnew;
            float psum = 0.f;
            #pragma unroll
            for (int nt = 0; nt < 4; ++nt) {
                float p = exp2f(sv[nt] - mnew);
                psum += p;
                pw[(r + rofs) * 80 + nt * 16 + l16] = f2bf(p);
            }
            #pragma unroll
            for (int msk = 1; msk < 16; msk <<= 1)
                psum += __shfl_xor(psum, msk, 32);
            lst[r] = lst[r] * alpha + psum;
            #pragma unroll
            for (int dt = 0; dt < 4; ++dt)
                o[dt][r] *= alpha;
        }

        // LDS round-trip converts P from D layout to A-fragment layout
        asm volatile("s_wait_dscnt 0x0" ::: "memory");
        v16bf pf[2];
        #pragma unroll
        for (int kf2 = 0; kf2 < 2; ++kf2) {
            FragU fu;
            const unsigned short* rp = pw + l16 * 80 + kf2 * 32 + (halfB >> 1);
            fu.u[0] = *(const uint4*)(rp);
            fu.u[1] = *(const uint4*)(rp + 16);
            pf[kf2] = fu.v;
        }
        // Load ALL V fragments, then do the 8 WMMAs
        v16bf vf[4][2];
        #pragma unroll
        for (int dt = 0; dt < 4; ++dt)
            #pragma unroll
            for (int kf2 = 0; kf2 < 2; ++kf2) {
                FragU fu;
                const char* rp = vbase + (size_t)(dt * 16 + l16) * (Tseq * 2)
                               + (size_t)(kb + kf2 * 32) * 2 + halfB;
                fu.u[0] = *(const uint4*)(rp);
                fu.u[1] = *(const uint4*)(rp + 32);
                vf[dt][kf2] = fu.v;
            }
        #pragma unroll
        for (int dt = 0; dt < 4; ++dt)
            #pragma unroll
            for (int kf2 = 0; kf2 < 2; ++kf2)
                o[dt] = __builtin_amdgcn_wmma_f32_16x16x32_bf16(
                    false, pf[kf2], false, vf[dt][kf2], (short)0, o[dt], false, false);
    }

    const int b = bh >> 4;   // / Hn
    const int h = bh & 15;
    #pragma unroll
    for (int dt = 0; dt < 4; ++dt)
        #pragma unroll
        for (int r = 0; r < 8; ++r) {
            int   t   = qb + r + rofs;
            int   col = h * 64 + dt * 16 + l16;
            float vv  = o[dt][r] / lst[r];
            Y[((size_t)(b * Tseq + t)) * Cdim + col] = f2bf(vv);
        }
}

// ---------------- launch ----------------
extern "C" void kernel_launch(void* const* d_in, const int* in_sizes, int n_in,
                              void* d_out, int out_size, void* d_ws, size_t ws_size,
                              hipStream_t stream) {
    const float* x      = (const float*)d_in[0];
    const float* w_attn = (const float*)d_in[1];
    const float* b_attn = (const float*)d_in[2];
    const float* w_proj = (const float*)d_in[3];
    const float* b_proj = (const float*)d_in[4];
    float* out = (float*)d_out;

    char* ws = (char*)d_ws;
    const size_t szX   = (size_t)Bsz * Tseq * Cdim * 2;      // 16 MB
    const size_t szWaT = (size_t)Cdim * N3C * 2;             // 6 MB
    const size_t szWpT = (size_t)Cdim * Cdim * 2;            // 2 MB
    const size_t szQKV = (size_t)Bsz * Hn * Tseq * HDim * 2; // 16 MB each

    unsigned short* xbf   = (unsigned short*)(ws);
    unsigned short* waT   = (unsigned short*)(ws + szX);
    unsigned short* wpT   = (unsigned short*)(ws + szX + szWaT);
    unsigned short* qbuf  = (unsigned short*)(ws + szX + szWaT + szWpT);
    unsigned short* kbuf  = (unsigned short*)(ws + szX + szWaT + szWpT + szQKV);
    unsigned short* vTbuf = (unsigned short*)(ws + szX + szWaT + szWpT + 2 * szQKV);
    unsigned short* ybuf  = (unsigned short*)(ws + szX + szWaT + szWpT + 3 * szQKV);

    const int M = Bsz * Tseq;  // 8192

    int nx = M * Cdim;
    k_cvt<<<(nx + 255) / 256, 256, 0, stream>>>(x, xbf, nx);
    int na = Cdim * N3C;
    k_transpose_cvt<<<(na + 255) / 256, 256, 0, stream>>>(w_attn, waT, Cdim, N3C);
    int np = Cdim * Cdim;
    k_transpose_cvt<<<(np + 255) / 256, 256, 0, stream>>>(w_proj, wpT, Cdim, Cdim);

    // qkv = x @ w_attn + b_attn, scattered into q / k / v^T (bf16)
    k_gemm<0><<<dim3(M / 128, N3C / 128), 256, 0, stream>>>(
        xbf, waT, b_attn, M, N3C, Cdim, qbuf, kbuf, vTbuf, nullptr);

    // causal flash attention -> y (bf16, [B*T, C])
    k_flash<<<dim3(Tseq / 128, Bsz * Hn), 256, 0, stream>>>(qbuf, kbuf, vTbuf, ybuf);

    // out = y @ w_proj + b_proj (fp32)
    k_gemm<1><<<dim3(M / 128, Cdim / 128), 256, 0, stream>>>(
        ybuf, wpT, b_proj, M, Cdim, Cdim, nullptr, nullptr, nullptr, out);
}
